// DeformConv2dBlock_71004399337612
// MI455X (gfx1250) — matrix operand order repaired and checked by reading, not compiled, not run on hardware
//
#include <hip/hip_runtime.h>
#include <hip/hip_bf16.h>

typedef _Float16 v16h __attribute__((ext_vector_type(16)));
typedef _Float16 v8h  __attribute__((ext_vector_type(8)));
typedef float    v8f  __attribute__((ext_vector_type(8)));
typedef float    v2f  __attribute__((ext_vector_type(2), aligned(4)));

#define BATCH 8
#define CHN   64
#define HH    128
#define WW    128
#define HWIMG (HH*WW)          // 16384
#define NPT   9
#define NOFF  18               // 2*NPT
#define KTOT  (CHN*NPT)        // 576, contraction index k = n*64 + c

// ---------------------------------------------------------------------------
// Kernel 0: convert w_conv (o, c, 3, 3) fp32 -> f16, permuted to (o, n, c)
// so B rows are contiguous in k = n*64 + c.
// ---------------------------------------------------------------------------
__global__ void cvt_wconv_kernel(const float* __restrict__ w,
                                 _Float16* __restrict__ wh, int ntot) {
    int i = blockIdx.x * 256 + threadIdx.x;
    if (i < ntot) {
        int o = i / KTOT; int r = i - o * KTOT;
        int n = r >> 6;   int c = r & 63;
        wh[i] = (_Float16)w[o * KTOT + c * NPT + n];
    }
}

// ---------------------------------------------------------------------------
// Kernel 1: offset conv (3x3, pad 1, C=64 -> 18) + bias; writes raw sampling
// positions p_x/p_y (zero-padded coords). Block = one 16x16 pixel tile.
// ---------------------------------------------------------------------------
__global__ __launch_bounds__(256) void offconv_kernel(
    const float* __restrict__ x, const float* __restrict__ w_off,
    const float* __restrict__ b_off,
    float* __restrict__ PX, float* __restrict__ PY)
{
    __shared__ float wof[NOFF * CHN * 9];   // 41472 B
    __shared__ float patch[16][18][18];     // 20736 B
    const int tid = threadIdx.x;
    const int blk = blockIdx.x;
    const int tw = blk & 7, th = (blk >> 3) & 7, bb = blk >> 6;
    const int h0 = th * 16, w0 = tw * 16;
    const int px_ = tid & 15, py_ = tid >> 4;
    const int h = h0 + py_, w = w0 + px_;

    for (int i = tid; i < NOFF * CHN * 9; i += 256) wof[i] = w_off[i];

    float acc[NOFF];
#pragma unroll
    for (int t = 0; t < NOFF; ++t) acc[t] = 0.f;

    const float* xb = x + (size_t)bb * CHN * HWIMG;
    for (int cc0 = 0; cc0 < CHN; cc0 += 16) {
        __syncthreads();
        for (int i = tid; i < 16 * 18 * 18; i += 256) {
            int c = i / 324; int r = i - c * 324;
            int ii = r / 18; int jj = r - ii * 18;
            int hg = h0 - 1 + ii, wg = w0 - 1 + jj;
            float v = 0.f;
            if (hg >= 0 && hg < HH && wg >= 0 && wg < WW)
                v = xb[(size_t)(cc0 + c) * HWIMG + hg * WW + wg];
            patch[c][ii][jj] = v;
        }
        __syncthreads();
#pragma unroll 1
        for (int c = 0; c < 16; ++c) {
            int cg = cc0 + c;
#pragma unroll
            for (int kh = 0; kh < 3; ++kh)
#pragma unroll
                for (int kw = 0; kw < 3; ++kw) {
                    float xv = patch[c][py_ + kh][px_ + kw];
#pragma unroll
                    for (int t = 0; t < NOFF; ++t)
                        acc[t] += xv * wof[(t * CHN + cg) * 9 + kh * 3 + kw];
                }
        }
    }

    const float pnx[NPT] = {-1,-1,-1, 0,0,0, 1,1,1};
    const float pny[NPT] = {-1, 0, 1,-1,0,1,-1,0,1};
    size_t pix = (size_t)bb * HWIMG + h * WW + w;
#pragma unroll
    for (int n = 0; n < NPT; ++n) {
        PX[pix * NPT + n] = acc[n]       + b_off[n]       + pnx[n] + (float)(h + 1);
        PY[pix * NPT + n] = acc[NPT + n] + b_off[NPT + n] + pny[n] + (float)(w + 1);
    }
}

// ---------------------------------------------------------------------------
// Kernel 2: fused bilinear-gather + GEMM via v_wmma_f32_16x16x32_f16.
// k = n*64 + c (n-major) -> each 32-wide K chunk has ONE sampling point n:
// bilinear params live in registers, gathers are paired b64 loads (2x2 block),
// zero LDS. One wave per 16-pixel tile; 4 accumulators = 64 output channels.
// ---------------------------------------------------------------------------
__global__ __launch_bounds__(256) void deform_wmma_kernel(
    const float* __restrict__ x, const _Float16* __restrict__ wh,
    const float* __restrict__ PX, const float* __restrict__ PY,
    float* __restrict__ out)
{
    const int lane = threadIdx.x & 31;
    const int wv   = threadIdx.x >> 5;
    const int tile = blockIdx.x * 8 + wv;
    const int pix_base  = tile * 16;          // flat pixel over (b,H,W)
    const int bb        = pix_base >> 14;     // /16384, tiles never cross images
    const int pimg_base = pix_base & 16383;

    const int m   = lane & 15;                // A row / B column select
    const int kba = (lane >= 16) ? 8  : 0;    // A K base (16-bit A 16x32 layout)
    const int kbb = (lane >= 16) ? 8  : 0;    // B K base (16-bit B 32x16 layout)
    const float* ximg = x + (size_t)bb * CHN * HWIMG;

    v8f acc[4] = {};

#pragma unroll 1
    for (int n = 0; n < NPT; ++n) {
        // --- bilinear params for (pixel m, point n), reference semantics ---
        size_t pidx = (size_t)(pix_base + m) * NPT + n;
        float px = PX[pidx], py = PY[pidx];
        float fx = floorf(px), fy = floorf(py);
        float qltx = fminf(fmaxf(fx,       0.f), 129.f);
        float qlty = fminf(fmaxf(fy,       0.f), 129.f);
        float qrbx = fminf(fmaxf(fx + 1.f, 0.f), 129.f);
        float qrby = fminf(fmaxf(fy + 1.f, 0.f), 129.f);
        float pxc  = fminf(fmaxf(px, 0.f), 129.f);
        float pyc  = fminf(fmaxf(py, 0.f), 129.f);
        float glt = (1.f + (qltx - pxc)) * (1.f + (qlty - pyc));
        float grb = (1.f - (qrbx - pxc)) * (1.f - (qrby - pyc));
        float glb = (1.f + (qltx - pxc)) * (1.f - (qrby - pyc));
        float grt = (1.f - (qrbx - pxc)) * (1.f + (qlty - pyc));
        int ix_lt = (int)qltx, ix_rb = (int)qrbx;
        int iy_lt = (int)qlty, iy_rb = (int)qrby;
        bool rA = (ix_lt >= 1) && (ix_lt <= HH);   // row q_lt_x interior?
        bool rB = (ix_rb >= 1) && (ix_rb <= HH);   // row q_rb_x interior?
        // base column: covers whichever of {q_lt_y, q_rb_y} is interior
        int by = min(max(iy_lt, 1), WW - 1);       // clamp to [1,127]
        // element<->corner weight matching (padding cols {0,129} never match)
        float wA0 = (rA && (by     == iy_lt) ? glt : 0.f)
                  + (rA && (by     == iy_rb) ? glb : 0.f);
        float wA1 = (rA && (by + 1 == iy_lt) ? glt : 0.f)
                  + (rA && (by + 1 == iy_rb) ? glb : 0.f);
        float wB0 = (rB && (by     == iy_lt) ? grt : 0.f)
                  + (rB && (by     == iy_rb) ? grb : 0.f);
        float wB1 = (rB && (by + 1 == iy_lt) ? grt : 0.f)
                  + (rB && (by + 1 == iy_rb) ? grb : 0.f);
        int rowA = min(max(ix_lt, 1), HH) - 1;     // clamped, weight-masked
        int rowB = min(max(ix_rb, 1), HH) - 1;
        int offA = rowA * WW + (by - 1);           // in [0, 16382]
        int offB = rowB * WW + (by - 1);

        const float* baseA = ximg + (size_t)kba * HWIMG + offA;
        const float* baseB = ximg + (size_t)kba * HWIMG + offB;

#pragma unroll
        for (int half = 0; half < 2; ++half) {     // channel blocks c0 = 0, 32
            const int k0 = n * 64 + half * 32;
            const float* pA = baseA + (size_t)(half * 32) * HWIMG;
            const float* pB = baseB + (size_t)(half * 32) * HWIMG;
            // A fragment: half j -> channel offset Koff beyond kba
            v16h a;
#pragma unroll
            for (int j = 0; j < 16; ++j) {
                const int Koff = (j < 8) ? j : j + 8;
                v2f a01 = *(const v2f*)(pA + (size_t)Koff * HWIMG);
                v2f b01 = *(const v2f*)(pB + (size_t)Koff * HWIMG);
                float v = wA0 * a01.x + wA1 * a01.y
                        + wB0 * b01.x + wB1 * b01.y;
                a[j] = (_Float16)v;
            }
            // B fragments: column o = t*16 + m, K = kbb + (j < 8 ? j : j + 8)
#pragma unroll
            for (int t = 0; t < 4; ++t) {
                const _Float16* bp = wh + (size_t)(t * 16 + m) * KTOT + k0 + kbb;
                v8h lo = *(const v8h*)bp;
                v8h hi = *(const v8h*)(bp + 16);
                v16h bfr = __builtin_shufflevector(lo, hi,
                    0,1,2,3,4,5,6,7,8,9,10,11,12,13,14,15);
                acc[t] = __builtin_amdgcn_wmma_f32_16x16x32_f16(
                    false, a, false, bfr, (short)0, acc[t], false, false);
            }
        }
    }

    // C/D layout: VGPR r, lanes 0-15 -> (M=r, N=lane); lanes 16-31 -> (M=r+8).
    const int madd = (lane >= 16) ? 8 : 0;
    const int ncol = lane & 15;
#pragma unroll
    for (int t = 0; t < 4; ++t) {
        int o = t * 16 + ncol;
#pragma unroll
        for (int r = 0; r < 8; ++r) {
            int mm = r + madd;
            size_t gi = ((size_t)(bb * CHN + o)) * HWIMG + (pimg_base + mm);
            out[gi] = acc[t][r] + x[gi];   // residual add
        }
    }
}

// ---------------------------------------------------------------------------
extern "C" void kernel_launch(void* const* d_in, const int* in_sizes, int n_in,
                              void* d_out, int out_size, void* d_ws, size_t ws_size,
                              hipStream_t stream) {
    const float* x      = (const float*)d_in[0];
    const float* w_off  = (const float*)d_in[1];
    const float* b_off  = (const float*)d_in[2];
    const float* w_conv = (const float*)d_in[3];
    float* out = (float*)d_out;

    const size_t npix = (size_t)BATCH * HWIMG;          // 131072
    float* PX = (float*)d_ws;                            // npix*9 floats
    float* PY = PX + npix * NPT;                         // npix*9 floats
    _Float16* WH = (_Float16*)(PY + npix * NPT);         // 64*576 halves

    cvt_wconv_kernel<<<(CHN * KTOT + 255) / 256, 256, 0, stream>>>(
        w_conv, WH, CHN * KTOT);
    offconv_kernel<<<BATCH * 8 * 8, 256, 0, stream>>>(x, w_off, b_off, PX, PY);
    deform_wmma_kernel<<<(int)(npix / 16 / 8), 256, 0, stream>>>(
        x, WH, PX, PY, out);
}
